// PrototypeAttentionMemory_35966056137357
// MI455X (gfx1250) — matrix operand order repaired and checked by reading, not compiled, not run on hardware
//
#include <hip/hip_runtime.h>
#include <hip/hip_bf16.h>

// ---------------------------------------------------------------------------
// PrototypeAttentionMemory fused kernel for MI455X (gfx1250, wave32, WMMA)
// B=16, Q=4096, D=256, P=400 (100 classes x 4)
// ---------------------------------------------------------------------------

typedef __attribute__((ext_vector_type(8)))  float   v8f;
typedef __attribute__((ext_vector_type(4)))  float   v4f;
typedef __attribute__((ext_vector_type(2)))  float   v2f;
typedef __attribute__((ext_vector_type(16))) __bf16  v16bf;
typedef __attribute__((ext_vector_type(8)))  __bf16  v8bf;

#define D_DIM   256
#define P_DIM   400
#define NCLASS  100
#define BQ      65536

#define LOG_STRIDE 404   // f32 stride for logits/d2 LDS (conflict-free 16 rows)
#define W_STRIDE   416   // bf16 stride for vhT (400 padded to 416)
#define M_STRIDE   264   // bf16 stride for transpose-staging LDS
// per-wave LDS: logits f32[16][404] + staging bf16[16][264] + 16 f32 scalars
#define WAVE_LDS   (16*LOG_STRIDE*4 + 16*M_STRIDE*2 + 16*4)   // 34368 B

// ws layout (bytes)
#define WS_PROTOS  0
#define WS_KH      204800
#define WS_VHT     409600      // [256][416] bf16
#define WS_WQ      622592
#define WS_WO      753664
#define WS_PSQ     884736

// d_out layout (floats): pooled | dist | class_dist | weights
#define OUT_POOLED  0
#define OUT_DIST    16777216
#define OUT_CDIST   16842752
#define OUT_WEIGHTS 23396352

static __device__ __forceinline__ __bf16 f2bf(float f) {
    unsigned u = __builtin_bit_cast(unsigned, f);
    u += 0x7fffu + ((u >> 16) & 1u);               // round-to-nearest-even
    return __builtin_bit_cast(__bf16, (unsigned short)(u >> 16));
}

static __device__ __forceinline__ v16bf combine8(v8bf lo, v8bf hi) {
    return __builtin_shufflevector(lo, hi, 0,1,2,3,4,5,6,7,8,9,10,11,12,13,14,15);
}

static __device__ __forceinline__ v8f wmma_bf16(v16bf a, v16bf b, v8f c) {
    // (neg_a, A, neg_b, B, c_mod, C, reuse_a, reuse_b)
    return __builtin_amdgcn_wmma_f32_16x16x32_bf16(false, a, false, b, (short)0, c,
                                                   false, false);
}

// Load the 8 B fragments of one 16x16 output tile (K = 256) as a batch.
static __device__ __forceinline__ void load_b8(const __bf16* __restrict__ row,
                                               int hi, v16bf b[8]) {
    #pragma unroll
    for (int c = 0; c < 8; ++c)
        b[c] = combine8(*(const v8bf*)(row + c * 32 + hi * 8), *(const v8bf*)(row + c * 32 + hi * 8 + 16));
}

// A-fragment from an LDS row (two 16B halves per lane)
static __device__ __forceinline__ v16bf load_a_lds(const __bf16* row, int c, int hi) {
    v8bf lo = *(const v8bf*)(row + c * 32 + hi * 8);
    v8bf h8 = *(const v8bf*)(row + c * 32 + hi * 8 + 16);
    return combine8(lo, h8);
}

// ---------------------------------------------------------------------------
// init: convert Wq/Wo to bf16, zero the vhT pad columns [400..415]
// ---------------------------------------------------------------------------
__global__ void pam_init_kernel(const float* __restrict__ Wq,
                                const float* __restrict__ Wo,
                                __bf16* __restrict__ Wq_bf,
                                __bf16* __restrict__ Wo_bf,
                                __bf16* __restrict__ vhT_bf) {
    int idx = blockIdx.x * blockDim.x + threadIdx.x;
    if (idx < D_DIM * D_DIM) {
        Wq_bf[idx] = f2bf(Wq[idx]);
        Wo_bf[idx] = f2bf(Wo[idx]);
    }
    if (idx < D_DIM * 16) {               // 256 rows x 16 pad cols
        int e = idx >> 4;
        int j = P_DIM + (idx & 15);
        vhT_bf[e * W_STRIDE + j] = f2bf(0.0f);
    }
}

// ---------------------------------------------------------------------------
// prep: protos = ema + offset; kh = protos*Wk^T; vhT = (protos*Wv^T)^T; p_sq
// ---------------------------------------------------------------------------
__global__ void pam_prep_kernel(const float* __restrict__ ema,
                                const float* __restrict__ off,
                                const float* __restrict__ Wk,
                                const float* __restrict__ Wv,
                                __bf16* __restrict__ protos_bf,
                                __bf16* __restrict__ kh_bf,
                                __bf16* __restrict__ vhT_bf,
                                float*  __restrict__ p_sq) {
    __shared__ float sp[D_DIM];
    __shared__ float red[D_DIM];
    const int p = blockIdx.x;
    const int t = threadIdx.x;
    float pr = ema[p * D_DIM + t] + off[p * D_DIM + t];
    sp[t]  = pr;
    red[t] = pr * pr;
    protos_bf[p * D_DIM + t] = f2bf(pr);
    __syncthreads();
    for (int s = 128; s > 0; s >>= 1) {
        if (t < s) red[t] += red[t + s];
        __syncthreads();
    }
    if (t == 0) p_sq[p] = red[0];
    float aK = 0.0f, aV = 0.0f;
    const float* wkr = Wk + t * D_DIM;
    const float* wvr = Wv + t * D_DIM;
    #pragma unroll 8
    for (int d = 0; d < D_DIM; ++d) {
        float x = sp[d];
        aK = fmaf(x, wkr[d], aK);
        aV = fmaf(x, wvr[d], aV);
    }
    kh_bf[p * D_DIM + t]     = f2bf(aK);
    vhT_bf[t * W_STRIDE + p] = f2bf(aV);
}

// ---------------------------------------------------------------------------
// fused main kernel: one wave == one 16-query tile; 4 waves / workgroup.
// Per-wave LDS = 34.4KB -> 137KB/WG -> 2 workgroups resident per WGP.
// Half-wave lane `hi` owns the interleaved 8-col groups {32k+8hi, 32k+16+8hi}
// of its row — identical to the bf16 A-fragment K-interleave, so softmax
// builds the weights A-fragments directly in registers (no LDS round-trip).
// ---------------------------------------------------------------------------
__global__ void pam_fused_kernel(const float* __restrict__ q,
                                 const __bf16* __restrict__ protos_bf,
                                 const __bf16* __restrict__ kh_bf,
                                 const __bf16* __restrict__ vhT_bf,
                                 const __bf16* __restrict__ Wq_bf,
                                 const __bf16* __restrict__ Wo_bf,
                                 const float*  __restrict__ p_sq,
                                 float* __restrict__ out_pooled,
                                 float* __restrict__ out_dist,
                                 float* __restrict__ out_cdist,
                                 float* __restrict__ out_weights) {
    extern __shared__ __align__(16) char smem[];
    const int lane = threadIdx.x & 31;
    const int wave = threadIdx.x >> 5;
    const int mrow = lane & 15;   // row (A-side) / N column (D-side) within tile
    const int hi   = lane >> 4;   // lane-half selector (K split / M+8 on D-side)
    const int row0 = (blockIdx.x * 4 + wave) * 16;   // base query row

    float*  s_buf = (float*)(smem + wave * WAVE_LDS);          // [16][LOG_STRIDE]
    __bf16* s_m   = (__bf16*)((char*)s_buf + 16 * LOG_STRIDE * 4);
    float*  s_sc  = (float*)((char*)s_m + 16 * M_STRIDE * 2);  // 16 scalars

    // ---- (a) load q tile -> bf16 A fragments; q_sq per row -----------------
    v16bf qa[8];
    {
        const float* qrow = q + (size_t)(row0 + mrow) * D_DIM;
        float qs = 0.0f;
        #pragma unroll
        for (int c = 0; c < 8; ++c) {
            const int k0 = c * 32 + hi * 8;
            v4f f0a = *(const v4f*)(qrow + k0);
            v4f f0b = *(const v4f*)(qrow + k0 + 4);
            v4f f1a = *(const v4f*)(qrow + k0 + 16);
            v4f f1b = *(const v4f*)(qrow + k0 + 20);
            #pragma unroll
            for (int j = 0; j < 4; ++j) {
                qa[c][j]      = f2bf(f0a[j]);
                qa[c][4 + j]  = f2bf(f0b[j]);
                qa[c][8 + j]  = f2bf(f1a[j]);
                qa[c][12 + j] = f2bf(f1b[j]);
                qs += f0a[j]*f0a[j] + f0b[j]*f0b[j] + f1a[j]*f1a[j] + f1b[j]*f1b[j];
            }
        }
        qs += __shfl_xor(qs, 16);
        if (hi == 0) s_sc[mrow] = qs;
    }
    __syncthreads();

    // ---- (b) distances: qp = q * protos^T -> d2 tiles staged in LDS --------
    {
        float qsM[8];
        #pragma unroll
        for (int r = 0; r < 8; ++r) qsM[r] = s_sc[r + 8 * hi];

        for (int nt = 0; nt < 25; ++nt) {
            v16bf bfrag[8];
            load_b8(protos_bf + (size_t)(nt * 16 + mrow) * D_DIM, hi, bfrag);
            float ps = p_sq[nt * 16 + mrow];
            v8f acc = {};
            #pragma unroll
            for (int c = 0; c < 8; ++c) acc = wmma_bf16(qa[c], bfrag[c], acc);
            #pragma unroll
            for (int r = 0; r < 8; ++r) {
                float d2 = fmaxf(qsM[r] + ps - 2.0f * acc[r], 0.0f);
                s_buf[(r + 8 * hi) * LOG_STRIDE + nt * 16 + mrow] = d2;
            }
        }
        __syncthreads();

        // row-major pass: per-class min (4 consecutive protos), running min
        const float* Drow = s_buf + mrow * LOG_STRIDE;
        float* crow = out_cdist + (size_t)(row0 + mrow) * NCLASS;
        float rmin = __builtin_inff();
        #pragma unroll
        for (int kc = 0; kc < 13; ++kc) {
            const int g0 = kc * 32 + 8 * hi;           // group of 8 = 2 classes
            v4f a = *(const v4f*)(Drow + g0);
            v4f b = *(const v4f*)(Drow + g0 + 4);
            float c0 = fminf(fminf(a[0], a[1]), fminf(a[2], a[3]));
            float c1 = fminf(fminf(b[0], b[1]), fminf(b[2], b[3]));
            rmin = fminf(rmin, fminf(c0, c1));
            v2f cd; cd[0] = sqrtf(c0); cd[1] = sqrtf(c1);
            __builtin_nontemporal_store(cd, (v2f*)(crow + kc * 8 + 2 * hi));
            if (kc < 12) {                              // second group (cols<400)
                const int g1 = g0 + 16;
                v4f e = *(const v4f*)(Drow + g1);
                v4f f = *(const v4f*)(Drow + g1 + 4);
                float c2 = fminf(fminf(e[0], e[1]), fminf(e[2], e[3]));
                float c3 = fminf(fminf(f[0], f[1]), fminf(f[2], f[3]));
                rmin = fminf(rmin, fminf(c2, c3));
                v2f cd2; cd2[0] = sqrtf(c2); cd2[1] = sqrtf(c3);
                __builtin_nontemporal_store(cd2, (v2f*)(crow + kc * 8 + 4 + 2 * hi));
            }
        }
        rmin = fminf(rmin, __shfl_xor(rmin, 16));
        if (hi == 0) out_dist[row0 + mrow] = sqrtf(rmin);
    }

    // ---- (c) qh = q * Wq^T, stage bf16 through LDS (D->A transpose) --------
    for (int nt = 0; nt < 16; ++nt) {
        v16bf bfrag[8];
        load_b8(Wq_bf + (size_t)(nt * 16 + mrow) * D_DIM, hi, bfrag);
        v8f acc = {};
        #pragma unroll
        for (int c = 0; c < 8; ++c) acc = wmma_bf16(qa[c], bfrag[c], acc);
        #pragma unroll
        for (int r = 0; r < 8; ++r)
            s_m[(r + 8 * hi) * M_STRIDE + nt * 16 + mrow] = f2bf(acc[r]);
    }
    __syncthreads();
    // qa is dead from here on.

    v16bf qha[8];
    #pragma unroll
    for (int c = 0; c < 8; ++c)
        qha[c] = load_a_lds(s_m + mrow * M_STRIDE, c, hi);
    __syncthreads();

    // ---- (d) logits = qh * kh^T * D^-0.5 -> LDS ----------------------------
    const float scale = 0.0625f;   // 256^-0.5
    for (int nt = 0; nt < 25; ++nt) {
        v16bf bfrag[8];
        load_b8(kh_bf + (size_t)(nt * 16 + mrow) * D_DIM, hi, bfrag);
        v8f acc = {};
        #pragma unroll
        for (int c = 0; c < 8; ++c) acc = wmma_bf16(qha[c], bfrag[c], acc);
        #pragma unroll
        for (int r = 0; r < 8; ++r)
            s_buf[(r + 8 * hi) * LOG_STRIDE + nt * 16 + mrow] = acc[r] * scale;
    }
    __syncthreads();
    // qha dead from here on.

    // ---- softmax over P=400, A-fragments built in registers ----------------
    v16bf wa[13];   // unnormalized exp weights, A-fragment layout
    {
        const float* Lrow = s_buf + mrow * LOG_STRIDE;
        float mx = -__builtin_inff();
        #pragma unroll
        for (int kc = 0; kc < 13; ++kc) {
            const int g0 = kc * 32 + 8 * hi;
            #pragma unroll
            for (int j = 0; j < 8; ++j) mx = fmaxf(mx, Lrow[g0 + j]);
            if (kc < 12) {
                #pragma unroll
                for (int j = 0; j < 8; ++j) mx = fmaxf(mx, Lrow[g0 + 16 + j]);
            }
        }
        mx = fmaxf(mx, __shfl_xor(mx, 16));

        float sum = 0.0f;
        #pragma unroll
        for (int kc = 0; kc < 13; ++kc) {
            const int g0 = kc * 32 + 8 * hi;
            #pragma unroll
            for (int j = 0; j < 8; ++j) {
                float e = __expf(Lrow[g0 + j] - mx);
                sum += e;
                wa[kc][j] = f2bf(e);
            }
            if (kc < 12) {
                #pragma unroll
                for (int j = 0; j < 8; ++j) {
                    float e = __expf(Lrow[g0 + 16 + j] - mx);
                    sum += e;
                    wa[kc][8 + j] = f2bf(e);
                }
            } else {
                #pragma unroll
                for (int j = 0; j < 8; ++j) wa[kc][8 + j] = f2bf(0.0f);
            }
        }
        sum += __shfl_xor(sum, 16);
        float inv = 1.0f / sum;
        if (hi == 0) s_sc[mrow] = inv;     // per-row 1/sum for phase (e)

        // normalized fp32 weights output (coalesced 32B stores)
        float* outw = out_weights + (size_t)(row0 + mrow) * P_DIM;
        #pragma unroll
        for (int kc = 0; kc < 13; ++kc) {
            const int g0 = kc * 32 + 8 * hi;
            v4f w0, w1;
            #pragma unroll
            for (int j = 0; j < 4; ++j) {
                w0[j] = __expf(Lrow[g0 + j] - mx) * inv;
                w1[j] = __expf(Lrow[g0 + 4 + j] - mx) * inv;
            }
            __builtin_nontemporal_store(w0, (v4f*)(outw + g0));
            __builtin_nontemporal_store(w1, (v4f*)(outw + g0 + 4));
            if (kc < 12) {
                v4f w2, w3;
                #pragma unroll
                for (int j = 0; j < 4; ++j) {
                    w2[j] = __expf(Lrow[g0 + 16 + j] - mx) * inv;
                    w3[j] = __expf(Lrow[g0 + 20 + j] - mx) * inv;
                }
                __builtin_nontemporal_store(w2, (v4f*)(outw + g0 + 16));
                __builtin_nontemporal_store(w3, (v4f*)(outw + g0 + 20));
            }
        }
    }
    __syncthreads();

    // ---- (e) pooled_pre = softmax(w) * vh, normalize by 1/sum --------------
    {
        float invM[8];
        #pragma unroll
        for (int r = 0; r < 8; ++r) invM[r] = s_sc[r + 8 * hi];

        for (int nt = 0; nt < 16; ++nt) {
            const __bf16* vrow = vhT_bf + (size_t)(nt * 16 + mrow) * W_STRIDE;
            v8f acc = {};
            #pragma unroll
            for (int g = 0; g < 3; ++g) {       // B fragments in groups of 4
                v16bf bfrag[4];
                #pragma unroll
                for (int u = 0; u < 4; ++u)
                    bfrag[u] = combine8(*(const v8bf*)(vrow + (g * 4 + u) * 32 + hi * 8), *(const v8bf*)(vrow + (g * 4 + u) * 32 + hi * 8 + 16));
                #pragma unroll
                for (int u = 0; u < 4; ++u)
                    acc = wmma_bf16(wa[g * 4 + u], bfrag[u], acc);
            }
            {
                v16bf blast = combine8(*(const v8bf*)(vrow + 12 * 32 + hi * 8), *(const v8bf*)(vrow + 12 * 32 + hi * 8 + 16));
                acc = wmma_bf16(wa[12], blast, acc);
            }
            #pragma unroll
            for (int r = 0; r < 8; ++r)
                s_m[(r + 8 * hi) * M_STRIDE + nt * 16 + mrow] = f2bf(acc[r] * invM[r]);
        }
    }
    __syncthreads();

    // ---- (f) pooled = pooled_pre * Wo^T ------------------------------------
    {
        v16bf pa[8];   // invariant A fragments (pooled_pre)
        #pragma unroll
        for (int c = 0; c < 8; ++c)
            pa[c] = load_a_lds(s_m + mrow * M_STRIDE, c, hi);

        for (int nt = 0; nt < 16; ++nt) {
            v16bf bfrag[8];
            load_b8(Wo_bf + (size_t)(nt * 16 + mrow) * D_DIM, hi, bfrag);
            v8f acc = {};
            #pragma unroll
            for (int c = 0; c < 8; ++c) acc = wmma_bf16(pa[c], bfrag[c], acc);
            #pragma unroll
            for (int r = 0; r < 8; ++r)
                __builtin_nontemporal_store(acc[r],
                    out_pooled + (size_t)(row0 + r + 8 * hi) * D_DIM + nt * 16 + mrow);
        }
    }
}

// ---------------------------------------------------------------------------
extern "C" void kernel_launch(void* const* d_in, const int* in_sizes, int n_in,
                              void* d_out, int out_size, void* d_ws, size_t ws_size,
                              hipStream_t stream) {
    const float* q   = (const float*)d_in[0];
    const float* ema = (const float*)d_in[1];
    const float* off = (const float*)d_in[2];
    const float* Wq  = (const float*)d_in[3];
    const float* Wk  = (const float*)d_in[4];
    const float* Wv  = (const float*)d_in[5];
    const float* Wo  = (const float*)d_in[6];

    char* ws = (char*)d_ws;
    __bf16* protos_bf = (__bf16*)(ws + WS_PROTOS);
    __bf16* kh_bf     = (__bf16*)(ws + WS_KH);
    __bf16* vhT_bf    = (__bf16*)(ws + WS_VHT);
    __bf16* Wq_bf     = (__bf16*)(ws + WS_WQ);
    __bf16* Wo_bf     = (__bf16*)(ws + WS_WO);
    float*  p_sq      = (float*)(ws + WS_PSQ);

    float* out         = (float*)d_out;
    float* out_pooled  = out + OUT_POOLED;
    float* out_dist    = out + OUT_DIST;
    float* out_cdist   = out + OUT_CDIST;
    float* out_weights = out + OUT_WEIGHTS;

    hipLaunchKernelGGL(pam_init_kernel, dim3(256), dim3(256), 0, stream,
                       Wq, Wo, Wq_bf, Wo_bf, vhT_bf);
    hipLaunchKernelGGL(pam_prep_kernel, dim3(400), dim3(256), 0, stream,
                       ema, off, Wk, Wv, protos_bf, kh_bf, vhT_bf, p_sq);
    hipLaunchKernelGGL(pam_fused_kernel, dim3(1024), dim3(128),
                       4 * WAVE_LDS, stream,
                       q, protos_bf, kh_bf, vhT_bf, Wq_bf, Wo_bf, p_sq,
                       out_pooled, out_dist, out_cdist, out_weights);
}
